// Net_45320494907927
// MI455X (gfx1250) — matrix operand order repaired and checked by reading, not compiled, not run on hardware
//
#include <hip/hip_runtime.h>

// ---------------------------------------------------------------------------
// Types for CDNA5 WMMA / TDM
// ---------------------------------------------------------------------------
typedef __bf16 bf16;
typedef __bf16 v16bf __attribute__((ext_vector_type(16)));
typedef float  v8f   __attribute__((ext_vector_type(8)));
typedef unsigned u32x4 __attribute__((ext_vector_type(4)));
// gcc-style vectors: match amdgcn builtin parameter types
typedef int      vs4i __attribute__((__vector_size__(16)));
typedef unsigned gu4  __attribute__((__vector_size__(16)));
typedef int      gi4  __attribute__((__vector_size__(16)));
typedef int      gi8  __attribute__((__vector_size__(32)));

// Optional gfx1250 async global->LDS path (compile-safe feature probe).
#if defined(__has_builtin)
#if __has_builtin(__builtin_amdgcn_global_load_async_to_lds_b128) && \
    __has_builtin(__builtin_amdgcn_s_wait_asynccnt)
#define USE_ASYNC_LDS 1
#endif
#if __has_builtin(__builtin_amdgcn_tensor_load_to_lds) && \
    __has_builtin(__builtin_amdgcn_s_wait_tensorcnt)
#define USE_TDM 1
#endif
#endif
#ifndef USE_ASYNC_LDS
#define USE_ASYNC_LDS 0
#endif
#ifndef USE_TDM
#define USE_TDM 0
#endif

#if USE_ASYNC_LDS
__device__ __forceinline__ void async_copy_b128(const void* gsrc, void* ldst) {
  __builtin_amdgcn_global_load_async_to_lds_b128(
      (__attribute__((address_space(1))) vs4i*)gsrc,
      (__attribute__((address_space(3))) vs4i*)ldst, 0, 0);
}
#endif

// ---------------------------------------------------------------------------
// Utility kernels
// ---------------------------------------------------------------------------
__global__ __launch_bounds__(256) void zero_u32_kernel(unsigned* __restrict__ p, long n) {
  long i = (long)blockIdx.x * blockDim.x + threadIdx.x;
  long stride = (long)gridDim.x * blockDim.x;
  for (; i < n; i += stride) p[i] = 0u;
}

__global__ __launch_bounds__(256) void scatter_kernel(const float* __restrict__ feats,
                                                      const int* __restrict__ coors,
                                                      bf16* __restrict__ grid,
                                                      unsigned char* __restrict__ mask,
                                                      int N, int C) {
  int i = blockIdx.x * blockDim.x + threadIdx.x;
  if (i >= N) return;
  int z = coors[i * 4 + 1], y = coors[i * 4 + 2], x = coors[i * 4 + 3];
  long v = ((long)z * 64 + y) * 64 + x;
  mask[v] = 1;
  for (int c = 0; c < C; ++c) grid[v * C + c] = (bf16)feats[(long)i * C + c];
}

__global__ __launch_bounds__(256) void count_mask_kernel(const unsigned char* __restrict__ mask,
                                                         float* __restrict__ cnt, int V) {
  int i = blockIdx.x * blockDim.x + threadIdx.x;
  if (i < V && mask[i]) atomicAdd(cnt, 1.0f);
}

__global__ __launch_bounds__(256) void pool_mask_kernel(const unsigned char* __restrict__ in,
                                                        unsigned char* __restrict__ out, int So) {
  int Si = So * 2;
  int V = So * So * So;
  int i = blockIdx.x * blockDim.x + threadIdx.x;
  if (i >= V) return;
  int z = i / (So * So), r = i - z * So * So, y = r / So, x = r - y * So;
  unsigned char a = 0;
  for (int dz = 0; dz < 2; ++dz)
    for (int dy = 0; dy < 2; ++dy)
      for (int dx = 0; dx < 2; ++dx)
        a |= in[((2 * z + dz) * Si + (2 * y + dy)) * Si + (2 * x + dx)];
  out[i] = a ? 1 : 0;
}

// Pack weights (kd,kh,kw,Cin,Cout) fp32 -> bf16 [Cout][Kpad], K = taps*Cin.
__global__ __launch_bounds__(256) void pack_w_kernel(const float* __restrict__ wsrc,
                                                     bf16* __restrict__ wt,
                                                     int Cin, int Cout, int K, int Kpad) {
  long idx = (long)blockIdx.x * blockDim.x + threadIdx.x;
  long total = (long)Cout * Kpad;
  if (idx >= total) return;
  int co = (int)(idx / Kpad);
  int k  = (int)(idx - (long)co * Kpad);
  float val = 0.0f;
  if (k < K) {
    int t = k / Cin, ci = k - t * Cin;
    val = wsrc[((long)t * Cin + ci) * Cout + co];
  }
  wt[idx] = (bf16)val;
}

// ---------------------------------------------------------------------------
// Implicit-GEMM 3D conv via V_WMMA_F32_16X16X32_BF16.
//   y[m][co] = sum_k A[m][k] * W[k][co],  k = tap*Cin + ci
// Block: 64 threads (2 wave32). M tile = 32 voxels; A staged to LDS with
// GLOBAL_LOAD_ASYNC_TO_LDS_B128 (ASYNCcnt). B panel (64 couts x 32 k) staged
// to LDS by the Tensor Data Mover (TENSOR_LOAD_TO_LDS, TENSORcnt) when
// available, else per-lane B128 loads. Each wave owns a 32x32 output tile:
// 4 accumulators, 4 WMMA per K step. K loop unrolled 2x.
// Requires: Cin % 16 == 0, Cout % 32 == 0, weights zero-padded to Kpad.
// ---------------------------------------------------------------------------
__global__ __launch_bounds__(64) void conv_wmma_kernel(
    const bf16* __restrict__ xin,   // [Sin^3][Cin] bf16
    const bf16* __restrict__ wt,    // [Cout][Kpad] bf16 (packed, zero-padded)
    float* __restrict__ yout,       // [Sout^3][Cout] f32
    int Sin, int Sout, int Cin, int Cout,
    int ksz, int stride, int pad, int K, int Kpad)
{
  __shared__ u32x4 ldsA[128];       // 32 rows x 32 bf16 = 2 KB
#if USE_TDM
  __shared__ u32x4 ldsB[256];       // 64 rows x 32 bf16 = 4 KB (TDM tile)
#endif
  const int tid   = threadIdx.x;
  const int lane  = tid & 31;
  const int hi    = lane >> 4;      // lane half -> K sub-range
  const int r     = lane & 15;      // row (A) / column (B,C,D)
  const int wv    = tid >> 5;
  const int mbase = blockIdx.x * 32;
  const int nblock0 = blockIdx.y * 64;
  const int nbase = nblock0 + wv * 32;
  const bool nvalid = (nbase < Cout);          // Cout % 32 == 0 -> uniform/wave
  const int M     = Sout * Sout * Sout;
  const int ksq   = ksz * ksz;

  // Staging role: each thread owns one row and one 16-element K half.
  const int srow  = tid >> 1;
  const int khalf = tid & 1;
  int sm = mbase + srow;
  if (sm >= M) sm = M - 1;          // clamp; zero-padded B keeps result exact
  int oz = sm / (Sout * Sout);  int mrem = sm - oz * Sout * Sout;
  int oy = mrem / Sout;         int ox = mrem - oy * Sout;
  const int izs = oz * stride - pad;
  const int iys = oy * stride - pad;
  const int ixs = ox * stride - pad;
  const int ldst = srow * 4 + khalf * 2;       // u32x4 slot for this thread

  v8f acc00 = {0.f,0.f,0.f,0.f,0.f,0.f,0.f,0.f};
  v8f acc01 = acc00, acc10 = acc00, acc11 = acc00;

#if USE_TDM
  const unsigned ldsBoff =
      (unsigned)(unsigned long long)(__attribute__((address_space(3))) void*)&ldsB[0];
  const int nloc0 = wv * 32 + r;    // block-local column row for B frags
#else
  const int nrow0 = nvalid ? (nbase + r)      : 0;
  const int nrow1 = nvalid ? (nbase + 16 + r) : 0;
  const u32x4* wtrow0 = (const u32x4*)(wt + (size_t)nrow0 * Kpad);
  const u32x4* wtrow1 = (const u32x4*)(wt + (size_t)nrow1 * Kpad);
#endif

#pragma unroll 2
  for (int k0 = 0; k0 < K; k0 += 32) {
    // ---- stage A tile (32 rows x 32 bf16) ----
    // A 16-aligned 16-span never straddles a tap (Cin % 16 == 0): one tap
    // decode per thread per chunk.
    int k  = k0 + khalf * 16;
    int t  = k / Cin;
    int ci = k - t * Cin;
    int dz = t / ksq;    int trem = t - dz * ksq;
    int dh = trem / ksz; int dwv  = trem - dh * ksz;
    int iz = izs + dz, iy = iys + dh, ix = ixs + dwv;
    bool inb = ((unsigned)iz < (unsigned)Sin) && ((unsigned)iy < (unsigned)Sin) &&
               ((unsigned)ix < (unsigned)Sin);
    if (inb) {
      const u32x4* src =
          (const u32x4*)(xin + ((size_t)((iz * Sin + iy) * Sin + ix)) * Cin + ci);
#if USE_ASYNC_LDS
      async_copy_b128(src,     &ldsA[ldst]);
      async_copy_b128(src + 1, &ldsA[ldst + 1]);
#else
      ldsA[ldst]     = src[0];
      ldsA[ldst + 1] = src[1];
#endif
    } else {
      u32x4 z4 = {0u, 0u, 0u, 0u};
      ldsA[ldst]     = z4;
      ldsA[ldst + 1] = z4;
    }

#if USE_TDM
    // ---- stage B panel via Tensor Data Mover (wave 0 only) ----
    if (wv == 0) {
      unsigned long long gaddr =
          (unsigned long long)(const void*)(wt + (size_t)nblock0 * Kpad + k0);
      unsigned td0 = (unsigned)(Kpad - k0);      // remaining K elements (dim0)
      unsigned td1 = (unsigned)(Cout - nblock0); // remaining rows: OOB -> 0
      unsigned long long s0 = (unsigned long long)(unsigned)Kpad;
      // D# group0: count=1 | lds_addr | global_addr | type=2 (bits 127:126)
      gu4 g0 = { 1u, ldsBoff,
                 (unsigned)(gaddr & 0xFFFFFFFFull),
                 (unsigned)((gaddr >> 32) & 0x01FFFFFFull) | 0x80000000u };
      // D# group1: wg_mask=0, data_size=1 (2B), tensor_dim0/1, tile 32x64,
      // dim0 stride = Kpad elements.
      gi8 g1 = { (int)0x10000u,
                 (int)((td0 & 0xFFFFu) << 16),
                 (int)(((td0 >> 16) & 0xFFFFu) | ((td1 & 0xFFFFu) << 16)),
                 (int)(((td1 >> 16) & 0xFFFFu) | (32u << 16)),
                 (int)64u,
                 (int)(unsigned)(s0 & 0xFFFFFFFFull),
                 (int)(unsigned)((s0 >> 32) & 0xFFFFull),
                 0 };
      gi4 gz4 = {0, 0, 0, 0};
      gi8 gz8 = {0, 0, 0, 0, 0, 0, 0, 0};
      __builtin_amdgcn_tensor_load_to_lds(g0, g1, gz4, gz4, gz8, 0);
      __builtin_amdgcn_s_wait_tensorcnt(0);
    }
#endif
#if USE_ASYNC_LDS
    __builtin_amdgcn_s_wait_asynccnt(0);
#endif
    __syncthreads();

    // ---- fragments ----
    union { v16bf v; u32x4 q[2]; } Af0, Af1, Bf0, Bf1;
    // A 16x32 layout: lanes 0-15 K 0..15; lanes 16-31 K 16..31
    Af0.q[0] = ldsA[r * 4 + hi * 2];
    Af0.q[1] = ldsA[r * 4 + hi * 2 + 1];
    Af1.q[0] = ldsA[(r + 16) * 4 + hi * 2];
    Af1.q[1] = ldsA[(r + 16) * 4 + hi * 2 + 1];
    // B 32x16 layout: lanes 0-15 K 0..15; lanes 16-31 K 16..31
#if USE_TDM
    const int bq2 = nloc0 * 4 + hi * 2;          // row-major 64x(16 pairs)
    Bf0.q[0] = ldsB[bq2];
    Bf0.q[1] = ldsB[bq2 + 1];
    Bf1.q[0] = ldsB[bq2 + 64];                   // +16 rows
    Bf1.q[1] = ldsB[bq2 + 64 + 1];
#else
    const int bq = (k0 >> 3) + hi * 2;
    Bf0.q[0] = wtrow0[bq];
    Bf0.q[1] = wtrow0[bq + 1];
    Bf1.q[0] = wtrow1[bq];
    Bf1.q[1] = wtrow1[bq + 1];
    __builtin_prefetch(wtrow0 + bq + 4, 0, 1);
    __builtin_prefetch(wtrow1 + bq + 4, 0, 1);
#endif

    acc00 = __builtin_amdgcn_wmma_f32_16x16x32_bf16(false, Af0.v, false, Bf0.v,
                                                    (short)0, acc00, false, false);
    acc10 = __builtin_amdgcn_wmma_f32_16x16x32_bf16(false, Af1.v, false, Bf0.v,
                                                    (short)0, acc10, false, false);
    acc01 = __builtin_amdgcn_wmma_f32_16x16x32_bf16(false, Af0.v, false, Bf1.v,
                                                    (short)0, acc01, false, false);
    acc11 = __builtin_amdgcn_wmma_f32_16x16x32_bf16(false, Af1.v, false, Bf1.v,
                                                    (short)0, acc11, false, false);
    __syncthreads();
  }

  // C/D layout: VGPR i -> row (i + 8*hi), lane r -> column.
  if (nvalid) {
#pragma unroll
    for (int i = 0; i < 8; ++i) {
      int m0 = mbase + i + hi * 8;
      if (m0 < M) {
        yout[(size_t)m0 * Cout + (nbase + r)]      = acc00[i];
        yout[(size_t)m0 * Cout + (nbase + 16 + r)] = acc01[i];
      }
      int m1 = mbase + 16 + i + hi * 8;
      if (m1 < M) {
        yout[(size_t)m1 * Cout + (nbase + r)]      = acc10[i];
        yout[(size_t)m1 * Cout + (nbase + 16 + r)] = acc11[i];
      }
    }
  }
}

// ---------------------------------------------------------------------------
// Masked BN: per-channel sum / sumsq over masked voxels (one-pass).
// ---------------------------------------------------------------------------
__global__ __launch_bounds__(32) void bn_stats_kernel(const float* __restrict__ y,
                                                      const unsigned char* __restrict__ mask,
                                                      float* __restrict__ sum,
                                                      float* __restrict__ sumsq,
                                                      int V, int C) {
  int c = blockIdx.y * 32 + threadIdx.x;
  float s = 0.f, s2 = 0.f;
  for (int v = blockIdx.x; v < V; v += gridDim.x) {
    if (mask[v]) {
      float t = y[(size_t)v * C + c];
      s += t; s2 += t * t;
    }
  }
  atomicAdd(&sum[c], s);
  atomicAdd(&sumsq[c], s2);
}

__global__ __launch_bounds__(256) void bn_apply_kernel(const float* __restrict__ y,
                                                       const unsigned char* __restrict__ mask,
                                                       const float* __restrict__ sum,
                                                       const float* __restrict__ sumsq,
                                                       const float* __restrict__ cnt,
                                                       const float* __restrict__ g,
                                                       const float* __restrict__ b,
                                                       bf16* __restrict__ outb,
                                                       float* __restrict__ outf,
                                                       int V, int C) {
  long idx = (long)blockIdx.x * blockDim.x + threadIdx.x;
  long total = (long)V * C;
  if (idx >= total) return;
  int  c = (int)(idx % C);
  long v = idx / C;
  float n    = fmaxf(cnt[0], 1.0f);
  float mean = sum[c] / n;
  float var  = sumsq[c] / n - mean * mean;
  float m    = mask[v] ? 1.0f : 0.0f;
  float xv   = y[idx];
  float rres = (xv - mean) * rsqrtf(var + 1e-5f) * g[c] + b[c];
  rres = fmaxf(rres, 0.0f) * m;
  outb[idx] = (bf16)rres;
  if (outf) outf[idx] = rres;
}

// ---------------------------------------------------------------------------
// conv_transpose 2x2x2 stride 2 VALID (tiny: 1^3->2^3, 2^3->4^3).
// out[2i+e] uses kernel tap (1-e) per dim (lhs-dilated correlation).
// ---------------------------------------------------------------------------
__global__ __launch_bounds__(256) void deconv_kernel(const bf16* __restrict__ xin,
                                                     const float* __restrict__ w,
                                                     float* __restrict__ yout,
                                                     int Sin, int Cin, int Cout) {
  int Sout = Sin * 2;
  int V = Sout * Sout * Sout;
  long idx = (long)blockIdx.x * blockDim.x + threadIdx.x;
  if (idx >= (long)V * Cout) return;
  int co = (int)(idx % Cout);
  int o  = (int)(idx / Cout);
  int oz = o / (Sout * Sout), r = o - oz * Sout * Sout;
  int oy = r / Sout, ox = r - oy * Sout;
  int t = ((1 - (oz & 1)) << 2) | ((1 - (oy & 1)) << 1) | (1 - (ox & 1));
  int i = ((oz >> 1) * Sin + (oy >> 1)) * Sin + (ox >> 1);
  float s = 0.f;
  for (int ci = 0; ci < Cin; ++ci)
    s += (float)xin[(size_t)i * Cin + ci] * w[((size_t)t * Cin + ci) * Cout + co];
  yout[idx] = s;
}

// ---------------------------------------------------------------------------
// Host orchestration
// ---------------------------------------------------------------------------
static void run_zero(void* p, long n_uints, hipStream_t s) {
  unsigned blocks = (unsigned)(((n_uints + 255) / 256) < 4096 ? ((n_uints + 255) / 256) : 4096);
  if (blocks == 0) blocks = 1;
  zero_u32_kernel<<<blocks, 256, 0, s>>>((unsigned*)p, n_uints);
}

static void run_conv(const bf16* xin, const float* wsrc, bf16* wt, float* yout,
                     int Sin, int Sout, int Cin, int Cout, int ksz, int stride, int pad,
                     hipStream_t s) {
  int taps = ksz * ksz * ksz;
  int K = taps * Cin;
  int Kpad = (K + 31) & ~31;
  long total = (long)Cout * Kpad;
  pack_w_kernel<<<(unsigned)((total + 255) / 256), 256, 0, s>>>(wsrc, wt, Cin, Cout, K, Kpad);
  int M = Sout * Sout * Sout;
  dim3 g((unsigned)((M + 31) / 32), (unsigned)((Cout + 63) / 64));
  conv_wmma_kernel<<<g, 64, 0, s>>>(xin, wt, yout, Sin, Sout, Cin, Cout, ksz, stride, pad, K, Kpad);
}

static void run_bn(const float* y, const unsigned char* mask, const float* cnt,
                   const float* g, const float* b, bf16* outb, float* outf,
                   int V, int C, float* sum, hipStream_t s) {
  run_zero(sum, 512, s);  // sum[256] + sumsq[256] contiguous
  dim3 gs((unsigned)(V < 512 ? V : 512), (unsigned)(C / 32));
  bn_stats_kernel<<<gs, 32, 0, s>>>(y, mask, sum, sum + 256, V, C);
  long total = (long)V * C;
  bn_apply_kernel<<<(unsigned)((total + 255) / 256), 256, 0, s>>>(
      y, mask, sum, sum + 256, cnt, g, b, outb, outf, V, C);
}

extern "C" void kernel_launch(void* const* d_in, const int* in_sizes, int n_in,
                              void* d_out, int out_size, void* d_ws, size_t ws_size,
                              hipStream_t stream) {
  (void)in_sizes; (void)n_in; (void)out_size; (void)ws_size;
  const int CH[7] = {64, 96, 128, 160, 192, 224, 256};

  const float* feats = (const float*)d_in[0];
  const int*   coors = (const int*)d_in[1];
  int pi = 2;
  const float *saw[7], *sag[7], *sab[7], *sbw[7], *sbg[7], *sbb[7];
  const float *dw[6], *dg[6], *db[6];
  for (int l = 0; l < 7; ++l) {
    saw[l] = (const float*)d_in[pi++]; sag[l] = (const float*)d_in[pi++]; sab[l] = (const float*)d_in[pi++];
    sbw[l] = (const float*)d_in[pi++]; sbg[l] = (const float*)d_in[pi++]; sbb[l] = (const float*)d_in[pi++];
    if (l < 6) { dw[l] = (const float*)d_in[pi++]; dg[l] = (const float*)d_in[pi++]; db[l] = (const float*)d_in[pi++]; }
  }
  const float* i0w = (const float*)d_in[pi++];
  const float* i0g = (const float*)d_in[pi++];
  const float* i0b = (const float*)d_in[pi++];
  const float* i1w = (const float*)d_in[pi++];
  const float* i1g = (const float*)d_in[pi++];
  const float* i1b = (const float*)d_in[pi++];

  // Workspace layout (256B aligned carve)
  char* wsp = (char*)d_ws;
  auto carve = [&](size_t bytes) { char* p = wsp; wsp += (bytes + 255) & ~(size_t)255; return p; };
  const size_t VOX64 = (size_t)64 * 64 * 64;
  bf16*  actA  = (bf16*)carve(VOX64 * 64 * 2);         // 32 MB
  bf16*  actB  = (bf16*)carve(VOX64 * 64 * 2);         // 32 MB
  float* convY = (float*)carve(VOX64 * 64 * 4);        // 64 MB
  bf16*  wtbuf = (bf16*)carve((size_t)6912 * 256 * 2); // 3.4 MB packed weights
  unsigned char* maskL[7];
  {
    unsigned char* mb = (unsigned char*)carve(320 * 1024);
    size_t mo = 0;
    for (int l = 0; l < 7; ++l) {
      int S = 64 >> l;
      maskL[l] = mb + mo;
      mo += (size_t)S * S * S;
    }
  }
  float* stats  = (float*)carve(4096);  // sum[256] | sumsq[256] | counts[8]
  float* counts = stats + 512;

  // --- init: zero level-0 grid (16ch), mask0, counts ---
  run_zero(actA, (long)(VOX64 * 16 * 2 / 4), stream);
  run_zero(maskL[0], (long)(VOX64 / 4), stream);
  run_zero(counts, 8, stream);

  const int N = 150000;
  scatter_kernel<<<(N + 255) / 256, 256, 0, stream>>>(feats, coors, actA, maskL[0], N, 16);
  count_mask_kernel<<<(unsigned)((VOX64 + 255) / 256), 256, 0, stream>>>(maskL[0], counts + 0, (int)VOX64);

  int S = 64, Cin = 16;
  bf16 *cur = actA, *oth = actB;
  for (int l = 0; l < 7; ++l) {
    int C = CH[l];
    int V = S * S * S;
    // subm block a
    run_conv(cur, saw[l], wtbuf, convY, S, S, Cin, C, 3, 1, 1, stream);
    run_bn(convY, maskL[l], counts + l, sag[l], sab[l], oth, nullptr, V, C, stats, stream);
    { bf16* t = cur; cur = oth; oth = t; }
    // subm block b
    run_conv(cur, sbw[l], wtbuf, convY, S, S, C, C, 3, 1, 1, stream);
    run_bn(convY, maskL[l], counts + l, sbg[l], sbb[l], oth, nullptr, V, C, stats, stream);
    { bf16* t = cur; cur = oth; oth = t; }
    if (l < 6) {
      int So = S / 2;
      int Vo = So * So * So;
      run_conv(cur, dw[l], wtbuf, convY, S, So, C, C, 2, 2, 0, stream);
      pool_mask_kernel<<<(unsigned)((Vo + 255) / 256), 256, 0, stream>>>(maskL[l], maskL[l + 1], So);
      count_mask_kernel<<<(unsigned)((Vo + 255) / 256), 256, 0, stream>>>(maskL[l + 1], counts + l + 1, Vo);
      run_bn(convY, maskL[l + 1], counts + l + 1, dg[l], db[l], oth, nullptr, Vo, C, stats, stream);
      { bf16* t = cur; cur = oth; oth = t; }
      S = So;
    }
    Cin = C;
  }

  // inverse conv 0: 1^3 x256 -> 2^3 x128, mask = masks[5] (2^3)
  deconv_kernel<<<(8 * 128 + 255) / 256, 256, 0, stream>>>(cur, i0w, convY, 1, 256, 128);
  run_bn(convY, maskL[5], counts + 5, i0g, i0b, oth, nullptr, 8, 128, stats, stream);
  { bf16* t = cur; cur = oth; oth = t; }
  // inverse conv 1: 2^3 x128 -> 4^3 x64, mask = masks[4] (4^3); final fp32 out
  deconv_kernel<<<(64 * 64 + 255) / 256, 256, 0, stream>>>(cur, i1w, convY, 2, 128, 64);
  run_bn(convY, maskL[4], counts + 4, i1g, i1b, oth, (float*)d_out, 64, 64, stats, stream);
}
